// RWKV_TimeMix_83056077570123
// MI455X (gfx1250) — compile-verified, matrix operand order checked by reading, not run on hardware
//
#include <hip/hip_runtime.h>
#include <hip/hip_bf16.h>
#include <stdint.h>

// ---------------------------------------------------------------------------
// RWKV time-mix for MI455X (gfx1250, wave32, WMMA)
//   B=8, T=768, C=1024  ->  M = B*T = 6144
// GEMMs in bf16 via v_wmma_f32_16x16x32_bf16 with double-buffered LDS and
// GLOBAL_LOAD_ASYNC_TO_LDS_B128 staging (ASYNCcnt pipeline); WKV "conv" done
// as a chunked linear-recurrence scan (3 passes, 65k-way parallel).
// ---------------------------------------------------------------------------

#ifndef __has_builtin
#define __has_builtin(x) 0
#endif
#if __has_builtin(__builtin_amdgcn_global_load_async_to_lds_b128) && \
    __has_builtin(__builtin_amdgcn_s_wait_asynccnt)
#define HAVE_ASYNC_LDS 1
#else
#define HAVE_ASYNC_LDS 0
#endif

typedef __attribute__((ext_vector_type(16))) __bf16 v16bf;
typedef __attribute__((ext_vector_type(8)))  float  v8f;
typedef __attribute__((ext_vector_type(4)))  int    v4i;

static __device__ __forceinline__ __bf16 to_bf16(float f) {
    unsigned u = __builtin_bit_cast(unsigned, f);
    unsigned r = u + 0x7FFFu + ((u >> 16) & 1u);   // round-to-nearest-even
    unsigned short h = (unsigned short)(r >> 16);
    return __builtin_bit_cast(__bf16, h);
}

// ---------------------------------------------------------------------------
// f32 -> bf16 conversion (weights)
// ---------------------------------------------------------------------------
__global__ __launch_bounds__(256) void cvt_bf16_kernel(
    const float* __restrict__ in, __bf16* __restrict__ out, int n)
{
    int i = blockIdx.x * 256 + threadIdx.x;
    if (i < n) out[i] = to_bf16(in[i]);
}

// ---------------------------------------------------------------------------
// xm = x * time_mix + shift(x) * (1 - time_mix), emitted in bf16
// ---------------------------------------------------------------------------
__global__ __launch_bounds__(256) void mix_kernel(
    const float* __restrict__ x, const float* __restrict__ tmix,
    __bf16* __restrict__ out, int T, int C, int total)
{
    int i = blockIdx.x * 256 + threadIdx.x;
    if (i >= total) return;
    int c = i % C;
    int t = (i / C) % T;
    float xv = x[i];
    float xs = (t == 0) ? 0.0f : x[i - C];
    float tm = tmix[c];
    out[i] = to_bf16(xv * tm + xs * (1.0f - tm));
}

// ---------------------------------------------------------------------------
// NT GEMM:  Out[m,n] = sum_k A[m,k] * W[n,k]     (A:[M,K] bf16, W:[N,K] bf16)
// 128x128 tile, K-tile 64, LDS double buffer. 8 waves: 4(M) x 2(N); each wave
// owns a 32x64 sub-tile = 2x4 wmma f32 accumulators.
// ---------------------------------------------------------------------------
#define TM 128
#define TN 128
#define TK 64
#define LDSTR 80   // bf16 per LDS row (160 B rows -> 32-B aligned fragments)

__global__ __launch_bounds__(256) void gemm_bf16_nt_kernel(
    const __bf16* __restrict__ A, const __bf16* __restrict__ W,
    float* __restrict__ Out, int M, int N, int K)
{
    __shared__ alignas(32) __bf16 As[2][TM * LDSTR];
    __shared__ alignas(32) __bf16 Ws[2][TN * LDSTR];

    const int tid  = threadIdx.x;
    const int lane = tid & 31;
    const int wave = tid >> 5;          // 0..7
    const int wm   = wave >> 1;         // 0..3 -> M offset wm*32
    const int wn   = wave & 1;          // 0..1 -> N offset wn*64
    const int lrow = lane & 15;
    const int lhf  = lane >> 4;

    const int blockM = blockIdx.y * TM;
    const int blockN = blockIdx.x * TN;

    // staging geometry: 1024 16-B chunks per tile, 4 per thread
    int srow[4], scol[4];
#pragma unroll
    for (int i = 0; i < 4; ++i) {
        int chunk = tid + 256 * i;
        srow[i] = chunk >> 3;           // 8 chunks per 64-col row
        scol[i] = (chunk & 7) * 8;      // bf16 column
    }

    v8f acc[2][4];
#pragma unroll
    for (int i = 0; i < 2; ++i)
#pragma unroll
        for (int j = 0; j < 4; ++j) acc[i][j] = (v8f){};

    auto do_mma = [&](const __bf16* as, const __bf16* ws) {
#pragma unroll
        for (int ks = 0; ks < TK; ks += 32) {
            v16bf afrag[2], bfrag[4];
#pragma unroll
            for (int mt = 0; mt < 2; ++mt)
                afrag[mt] = *reinterpret_cast<const v16bf*>(
                    &as[(wm * 32 + mt * 16 + lrow) * LDSTR + ks + 16 * lhf]);
#pragma unroll
            for (int nt = 0; nt < 4; ++nt)
                bfrag[nt] = *reinterpret_cast<const v16bf*>(
                    &ws[(wn * 64 + nt * 16 + lrow) * LDSTR + ks + 16 * lhf]);
#pragma unroll
            for (int mt = 0; mt < 2; ++mt)
#pragma unroll
                for (int nt = 0; nt < 4; ++nt)
                    acc[mt][nt] = __builtin_amdgcn_wmma_f32_16x16x32_bf16(
                        false, afrag[mt], false, bfrag[nt],
                        (short)0, acc[mt][nt], false, false);
        }
    };

    const int ntiles = K / TK;

#if HAVE_ASYNC_LDS
    // ---- async global->LDS pipeline (ASYNCcnt), 8 b128 per thread/tile ----
    auto issue = [&](int tile, int buf) {
        int k0 = tile * TK;
#pragma unroll
        for (int i = 0; i < 4; ++i) {
            const __bf16* ga = &A[(size_t)(blockM + srow[i]) * K + k0 + scol[i]];
            const __bf16* gw = &W[(size_t)(blockN + srow[i]) * K + k0 + scol[i]];
            __builtin_amdgcn_global_load_async_to_lds_b128(
                (v4i*)ga, (v4i*)&As[buf][srow[i] * LDSTR + scol[i]], 0, 0);
            __builtin_amdgcn_global_load_async_to_lds_b128(
                (v4i*)gw, (v4i*)&Ws[buf][srow[i] * LDSTR + scol[i]], 0, 0);
        }
    };
    issue(0, 0);
    for (int t = 0; t < ntiles; ++t) {
        int cur = t & 1;
        if (t + 1 < ntiles) {
            issue(t + 1, cur ^ 1);
            __builtin_amdgcn_s_wait_asynccnt(8);   // current tile landed
        } else {
            __builtin_amdgcn_s_wait_asynccnt(0);
        }
        __syncthreads();
        do_mma(As[cur], Ws[cur]);
        __syncthreads();                           // reads done before reuse
    }
#else
    // ---- register-staged double buffer (one barrier per K-tile) ----------
    float4 ra[4], rw[4];
    auto gload = [&](int k0) {
#pragma unroll
        for (int i = 0; i < 4; ++i) {
            size_t aoff = (size_t)(blockM + srow[i]) * K + k0 + scol[i];
            size_t woff = (size_t)(blockN + srow[i]) * K + k0 + scol[i];
            ra[i] = *reinterpret_cast<const float4*>(&A[aoff]);
            rw[i] = *reinterpret_cast<const float4*>(&W[woff]);
            __builtin_prefetch(&A[aoff + TK], 0, 0);
            __builtin_prefetch(&W[woff + TK], 0, 0);
        }
    };
    auto sstore = [&](int buf) {
#pragma unroll
        for (int i = 0; i < 4; ++i) {
            *reinterpret_cast<float4*>(&As[buf][srow[i] * LDSTR + scol[i]]) = ra[i];
            *reinterpret_cast<float4*>(&Ws[buf][srow[i] * LDSTR + scol[i]]) = rw[i];
        }
    };
    gload(0);
    sstore(0);
    for (int t = 0; t < ntiles; ++t) {
        int cur = t & 1;
        if (t + 1 < ntiles) gload((t + 1) * TK);   // in flight across barrier
        __syncthreads();
        do_mma(As[cur], Ws[cur]);
        if (t + 1 < ntiles) sstore(cur ^ 1);
    }
#endif

    // ---- epilogue: f32 C/D layout: lanes 0-15 -> M=j, 16-31 -> M=j+8 ------
#pragma unroll
    for (int mt = 0; mt < 2; ++mt)
#pragma unroll
        for (int nt = 0; nt < 4; ++nt) {
            int col = blockN + wn * 64 + nt * 16 + lrow;
#pragma unroll
            for (int j = 0; j < 8; ++j) {
                int row = blockM + wm * 32 + mt * 16 + 8 * lhf + j;
                Out[(size_t)row * N + col] = acc[mt][nt][j];
            }
        }
}

// ---------------------------------------------------------------------------
// WKV recurrence as a chunked linear scan.
//   u_t = kv_t (num) / k_t (den);  S_{t+1} = a*S_t + u_t,  a = exp(-exp(decay))
//   wkv_t = exp(first)*u_t + S_t
// Pass 1: per-chunk Horner partial P_j = sum a^{L-1-i} u_i      (B*C*NCH thr)
// Pass 2: sequential combine over NCH chunks, stores chunk-start states
// Pass 3: replay chunk from start state, gate with sigmoid(r), emit bf16
// ---------------------------------------------------------------------------
#define NCH 8

__global__ __launch_bounds__(256) void wkv_partial_kernel(
    const float* __restrict__ Kp, const float* __restrict__ Vp,
    float* __restrict__ Pnum, float* __restrict__ Pden,
    const float* __restrict__ decay, int T, int C)
{
    int c = blockIdx.x * 256 + threadIdx.x;
    int j = blockIdx.y;           // chunk
    int b = blockIdx.z;
    if (c >= C) return;
    const int L = T / NCH;
    const float a = __expf(-__expf(decay[c]));
    float pn = 0.0f, pd = 0.0f;
    size_t base = ((size_t)b * T + (size_t)j * L) * C + c;
    for (int i = 0; i < L; ++i) {
        size_t idx = base + (size_t)i * C;
        float ek  = __expf(fminf(Kp[idx], 60.0f));
        float ekv = ek * Vp[idx];
        pn = a * pn + ekv;
        pd = a * pd + ek;
    }
    size_t pidx = ((size_t)b * NCH + j) * C + c;
    Pnum[pidx] = pn;
    Pden[pidx] = pd;
}

__global__ __launch_bounds__(256) void wkv_scan_kernel(
    float* __restrict__ Pnum, float* __restrict__ Pden,
    const float* __restrict__ decay, int T, int C)
{
    int c = blockIdx.x * 256 + threadIdx.x;
    int b = blockIdx.y;
    if (c >= C) return;
    const int L = T / NCH;
    const float d  = __expf(decay[c]);
    const float aL = __expf(-d * (float)L);       // a^L exactly
    float sn = 0.0f, sd = 0.0f;
    for (int j = 0; j < NCH; ++j) {
        size_t pidx = ((size_t)b * NCH + j) * C + c;
        float pn = Pnum[pidx], pd = Pden[pidx];
        Pnum[pidx] = sn;                          // in-place: chunk-start state
        Pden[pidx] = sd;
        sn = aL * sn + pn;
        sd = aL * sd + pd;
    }
}

__global__ __launch_bounds__(256) void wkv_final_kernel(
    const float* __restrict__ Kp, const float* __restrict__ Vp,
    const float* __restrict__ Rp,
    const float* __restrict__ Pnum, const float* __restrict__ Pden,
    const float* __restrict__ decay, const float* __restrict__ first,
    __bf16* __restrict__ out, int T, int C)
{
    int c = blockIdx.x * 256 + threadIdx.x;
    int j = blockIdx.y;
    int b = blockIdx.z;
    if (c >= C) return;
    const int L = T / NCH;
    const float a  = __expf(-__expf(decay[c]));
    const float ef = __expf(first[c]);
    size_t pidx = ((size_t)b * NCH + j) * C + c;
    float sn = Pnum[pidx], sd = Pden[pidx];
    size_t base = ((size_t)b * T + (size_t)j * L) * C + c;
    for (int i = 0; i < L; ++i) {
        size_t idx = base + (size_t)i * C;
        float ek  = __expf(fminf(Kp[idx], 60.0f));
        float ekv = ek * Vp[idx];
        float num = ef * ekv + sn;
        float den = ef * ek + sd + 1e-9f;
        float sig = 1.0f / (1.0f + __expf(-Rp[idx]));
        out[idx] = to_bf16(sig * num / den);
        sn = a * sn + ekv;
        sd = a * sd + ek;
    }
}

// ---------------------------------------------------------------------------
// Host launcher
// ---------------------------------------------------------------------------
extern "C" void kernel_launch(void* const* d_in, const int* in_sizes, int n_in,
                              void* d_out, int out_size, void* d_ws, size_t ws_size,
                              hipStream_t stream)
{
    const float* x          = (const float*)d_in[0];
    const float* time_decay = (const float*)d_in[1];
    const float* time_first = (const float*)d_in[2];
    const float* time_mix   = (const float*)d_in[3];
    const float* Wk         = (const float*)d_in[4];
    const float* Wv         = (const float*)d_in[5];
    const float* Wr         = (const float*)d_in[6];
    const float* Wo         = (const float*)d_in[7];

    const int C = in_sizes[3];              // time_mix has C elements
    const int T = 768;
    const int M = in_sizes[0] / C;          // B*T
    const int B = M / T;

    // ---- workspace carve-up (256-B aligned) ------------------------------
    char* ws = (char*)d_ws;
    size_t off = 0;
    auto carve = [&](size_t bytes) -> char* {
        char* p = ws + off;
        off += (bytes + 255) & ~(size_t)255;
        return p;
    };
    __bf16* Axm   = (__bf16*)carve((size_t)M * C * sizeof(__bf16));
    __bf16* RWKVb = (__bf16*)carve((size_t)M * C * sizeof(__bf16));
    __bf16* Wkb   = (__bf16*)carve((size_t)C * C * sizeof(__bf16));
    __bf16* Wvb   = (__bf16*)carve((size_t)C * C * sizeof(__bf16));
    __bf16* Wrb   = (__bf16*)carve((size_t)C * C * sizeof(__bf16));
    __bf16* Wob   = (__bf16*)carve((size_t)C * C * sizeof(__bf16));
    float*  Kf    = (float*)carve((size_t)M * C * sizeof(float));
    float*  Vf    = (float*)carve((size_t)M * C * sizeof(float));
    float*  Rf    = (float*)carve((size_t)M * C * sizeof(float));
    float*  Pnum  = (float*)carve((size_t)B * NCH * C * sizeof(float));
    float*  Pden  = (float*)carve((size_t)B * NCH * C * sizeof(float));

    // ---- 1) weights -> bf16 ---------------------------------------------
    {
        int n = C * C;
        int g = (n + 255) / 256;
        cvt_bf16_kernel<<<g, 256, 0, stream>>>(Wk, Wkb, n);
        cvt_bf16_kernel<<<g, 256, 0, stream>>>(Wv, Wvb, n);
        cvt_bf16_kernel<<<g, 256, 0, stream>>>(Wr, Wrb, n);
        cvt_bf16_kernel<<<g, 256, 0, stream>>>(Wo, Wob, n);
    }

    // ---- 2) time-shift mix -> bf16 --------------------------------------
    {
        int total = M * C;
        mix_kernel<<<(total + 255) / 256, 256, 0, stream>>>(
            x, time_mix, Axm, T, C, total);
    }

    // ---- 3) K/V/R projections (WMMA GEMMs) -------------------------------
    {
        dim3 grid(C / TN, M / TM);
        gemm_bf16_nt_kernel<<<grid, 256, 0, stream>>>(Axm, Wkb, Kf, M, C, C);
        gemm_bf16_nt_kernel<<<grid, 256, 0, stream>>>(Axm, Wvb, Vf, M, C, C);
        gemm_bf16_nt_kernel<<<grid, 256, 0, stream>>>(Axm, Wrb, Rf, M, C, C);
    }

    // ---- 4) WKV chunked scan + gate --------------------------------------
    {
        dim3 g1((C + 255) / 256, NCH, B);
        wkv_partial_kernel<<<g1, 256, 0, stream>>>(Kf, Vf, Pnum, Pden,
                                                   time_decay, T, C);
        dim3 g2((C + 255) / 256, B);
        wkv_scan_kernel<<<g2, 256, 0, stream>>>(Pnum, Pden, time_decay, T, C);
        wkv_final_kernel<<<g1, 256, 0, stream>>>(Kf, Vf, Rf, Pnum, Pden,
                                                 time_decay, time_first,
                                                 RWKVb, T, C);
    }

    // ---- 5) output projection -> f32 d_out -------------------------------
    {
        dim3 grid(C / TN, M / TM);
        gemm_bf16_nt_kernel<<<grid, 256, 0, stream>>>(
            RWKVb, Wob, (float*)d_out, M, C, C);
    }
}
